// Attention_16716012716212
// MI455X (gfx1250) — matrix operand order repaired and checked by reading, not compiled, not run on hardware
//
#include <hip/hip_runtime.h>

// ---------------------------------------------------------------------------
// Problem constants (match reference)
// ---------------------------------------------------------------------------
#define DIMV      4096
#define N_HEADS   32
#define N_KV      8
#define HEAD_DIM  128
#define BSZ       2
#define SEQ       2048
#define MTOK      ((size_t)BSZ * SEQ)        // 4096 token rows
#define KVDIM     (N_KV * HEAD_DIM)          // 1024

typedef float v8f   __attribute__((ext_vector_type(8)));
typedef __bf16 bf16x16 __attribute__((ext_vector_type(16)));
typedef int v4i_vs __attribute__((vector_size(16)));   // matches builtin prototype

#define GAS __attribute__((address_space(1)))
#define LAS __attribute__((address_space(3)))

// ---------------------------------------------------------------------------
// CDNA5 async global->LDS copy (ASYNCcnt-tracked, no VGPR round trip).
// Falls back to a synchronous VGPR copy if the builtin is unavailable.
// ---------------------------------------------------------------------------
#if __has_builtin(__builtin_amdgcn_global_load_async_to_lds_b128)
#define HAVE_ASYNC_LDS 1
#else
#define HAVE_ASYNC_LDS 0
#endif

__device__ __forceinline__ void async_cp16(const __bf16* g, __bf16* l) {
#if HAVE_ASYNC_LDS
  __builtin_amdgcn_global_load_async_to_lds_b128(
      (GAS v4i_vs*)g, (LAS v4i_vs*)l, 0, 0);
#else
  *(uint4*)l = *(const uint4*)g;
#endif
}

template <int N>
__device__ __forceinline__ void wait_async() {
#if HAVE_ASYNC_LDS
#if __has_builtin(__builtin_amdgcn_s_wait_asynccnt)
  __builtin_amdgcn_s_wait_asynccnt(N);
#else
  asm volatile("s_wait_asynccnt %0" ::"n"(N) : "memory");
#endif
#endif
}

// ---------------------------------------------------------------------------
// WMMA fragment loaders — this code's contiguous k order k = 16*(lane>>4) + e, not the ISA table order (wave32)
// ---------------------------------------------------------------------------
// A-matrix 16x32 (MxK), 16-bit: lanes 0-15 -> M=lane, K 0..15; lanes 16-31 ->
// M=lane-16, K 16..31.  VGPR v: K = kb+2v.
__device__ __forceinline__ bf16x16 frag_a(const __bf16* p, int ld, int lane) {
  const int m  = lane & 15;
  const int kb = (lane & 16) ? 16 : 0;
  bf16x16 f;
#pragma unroll
  for (int v = 0; v < 8; ++v) {
    int k = kb + 2 * v;
    f[2 * v]     = p[m * ld + k];
    f[2 * v + 1] = p[m * ld + k + 1];
  }
  return f;
}

// B-matrix 32x16 (KxN), 16-bit, source stored N-major [n][k] (stride ld):
// lane 0-15 -> N=lane, K 0..15 ; lane 16-31 -> N=lane-16, K 16..31.
__device__ __forceinline__ bf16x16 frag_b(const __bf16* p, int ld, int lane) {
  const int n    = lane & 15;
  const int koff = (lane & 16) ? 16 : 0;
  bf16x16 f;
#pragma unroll
  for (int e = 0; e < 16; ++e) f[e] = p[n * ld + koff + e];
  return f;
}

// B-matrix 32x16 where source is stored K-major [k][n] (stride ld) — used for
// P @ V where V tiles are [key][d].
__device__ __forceinline__ bf16x16 frag_b_kmajor(const __bf16* p, int ld, int lane) {
  const int n    = lane & 15;
  const int koff = (lane & 16) ? 16 : 0;
  bf16x16 f;
#pragma unroll
  for (int e = 0; e < 16; ++e) f[e] = p[(koff + e) * ld + n];
  return f;
}

// ---------------------------------------------------------------------------
// f32 -> bf16 downconvert (4 elems / thread; n multiple of 1024)
// ---------------------------------------------------------------------------
__global__ __launch_bounds__(256) void cvt_f32_bf16(const float* __restrict__ in,
                                                    __bf16* __restrict__ out,
                                                    size_t n) {
  size_t i = ((size_t)blockIdx.x * 256 + threadIdx.x) * 4;
  if (i + 3 < n) {
    float4 v = *(const float4*)(in + i);
    out[i + 0] = (__bf16)v.x;
    out[i + 1] = (__bf16)v.y;
    out[i + 2] = (__bf16)v.z;
    out[i + 3] = (__bf16)v.w;
  }
}

// ---------------------------------------------------------------------------
// GEMM: C[M][N] f32 = A[M][K]bf16 * B[N][K]bf16^T  (both operands K-contig)
// 128x128 tile / workgroup, 8 waves, K-step 32.
// Double-buffered async global->LDS staging: prefetch stage k+1 while WMMAs
// consume stage k.  4 async ops/thread/stage -> s_wait_asynccnt 4 after the
// prefetch issue guarantees stage k landed (async loads complete in order).
// ---------------------------------------------------------------------------
__global__ __launch_bounds__(256) void gemm_bf16_f32(const __bf16* __restrict__ A,
                                                     const __bf16* __restrict__ B,
                                                     float* __restrict__ C,
                                                     int M, int N, int K) {
  __shared__ __align__(16) __bf16 As[2][128][32];
  __shared__ __align__(16) __bf16 Bs[2][128][32];

  const int tid  = threadIdx.x;
  const int lane = tid & 31;
  const int wave = tid >> 5;
  const int m0   = blockIdx.y * 128;
  const int n0   = blockIdx.x * 128;
  const int nk   = K >> 5;

  v8f acc[8];
#pragma unroll
  for (int i = 0; i < 8; ++i) acc[i] = (v8f){0.f, 0.f, 0.f, 0.f, 0.f, 0.f, 0.f, 0.f};

  auto stage = [&](int buf, int k0) {
#pragma unroll
    for (int i = 0; i < 2; ++i) {
      int idx = tid + i * 256;      // 0..511 -> 128 rows x 4 x 16B segments
      int row = idx >> 2;
      int seg = idx & 3;
      async_cp16(A + (size_t)(m0 + row) * K + k0 + seg * 8, &As[buf][row][seg * 8]);
      async_cp16(B + (size_t)(n0 + row) * K + k0 + seg * 8, &Bs[buf][row][seg * 8]);
    }
  };

  stage(0, 0);
  for (int kt = 0; kt < nk; ++kt) {
    const int cur = kt & 1;
    if (kt + 1 < nk) {
      stage(cur ^ 1, (kt + 1) << 5);   // prefetch next stage into other buffer
      wait_async<4>();                 // previous stage's 4 ops complete
    } else {
      wait_async<0>();
    }
    __syncthreads();                   // publish stage kt to all waves

    bf16x16 af = frag_a(&As[cur][wave * 16][0], 32, lane);
#pragma unroll
    for (int nt = 0; nt < 8; ++nt) {
      bf16x16 bfr = frag_b(&Bs[cur][nt * 16][0], 32, lane);
      acc[nt] = __builtin_amdgcn_wmma_f32_16x16x32_bf16(
          false, af, false, bfr, (short)0, acc[nt], false, false);
    }
    __syncthreads();                   // everyone done reading buf `cur`
  }

  // C fragment layout: VGPR r -> row = r + 8*(lane>>4), col = lane&15.
  const int cn   = lane & 15;
  const int half = lane >> 4;
#pragma unroll
  for (int nt = 0; nt < 8; ++nt)
#pragma unroll
    for (int r = 0; r < 8; ++r) {
      int m = wave * 16 + r + half * 8;
      C[(size_t)(m0 + m) * N + n0 + nt * 16 + cn] = acc[nt][r];
    }
}

// ---------------------------------------------------------------------------
// RoPE + layout pack: f32 [b*s][h*128] -> bf16 [b][h][s][128]
// ---------------------------------------------------------------------------
__global__ __launch_bounds__(256) void rope_pack(const float* __restrict__ X,
                                                 const float* __restrict__ cosf,
                                                 const float* __restrict__ sinf,
                                                 __bf16* __restrict__ out,
                                                 int nheads) {
  size_t idx = (size_t)blockIdx.x * 256 + threadIdx.x;
  int i = (int)(idx & 63);
  size_t t = idx >> 6;
  int h = (int)(t % nheads); t /= nheads;
  int s = (int)(t % SEQ);
  int b = (int)(t / SEQ);
  if (b >= BSZ) return;

  float c  = cosf[s * 64 + i];
  float sn = sinf[s * 64 + i];
  size_t src = ((size_t)b * SEQ + s) * ((size_t)nheads * HEAD_DIM) + (size_t)h * HEAD_DIM + 2 * i;
  float te = X[src], to = X[src + 1];
  size_t dst = (((size_t)b * nheads + h) * SEQ + s) * HEAD_DIM + 2 * i;
  out[dst]     = (__bf16)(te * c - to * sn);
  out[dst + 1] = (__bf16)(te * sn + to * c);
}

// V pack: f32 [b*s][kv*128] -> bf16 [b][kv][s][128]
__global__ __launch_bounds__(256) void v_pack(const float* __restrict__ V,
                                              __bf16* __restrict__ out) {
  size_t idx = (size_t)blockIdx.x * 256 + threadIdx.x;
  int d = (int)(idx & (HEAD_DIM - 1));
  size_t t = idx >> 7;
  int kv = (int)(t % N_KV); t /= N_KV;
  int s = (int)(t % SEQ);
  int b = (int)(t / SEQ);
  if (b >= BSZ) return;
  size_t src = ((size_t)b * SEQ + s) * KVDIM + (size_t)kv * HEAD_DIM + d;
  size_t dst = (((size_t)b * N_KV + kv) * SEQ + s) * HEAD_DIM + d;
  out[dst] = (__bf16)V[src];
}

// ---------------------------------------------------------------------------
// Fused causal GQA flash attention.
// Grid: (SEQ/64, N_HEADS, BSZ).  Block: 128 threads = 4 waves.
// Wave w owns query rows [w*16, w*16+16).  K/V blocks are double-buffered and
// staged with async global->LDS copies (16 ops/thread/stage).
// ---------------------------------------------------------------------------
__global__ __launch_bounds__(128) void flash_attn(const __bf16* __restrict__ Q,  // [B][H][S][D]
                                                  const __bf16* __restrict__ Kk, // [B][KV][S][D]
                                                  const __bf16* __restrict__ Vv, // [B][KV][S][D]
                                                  __bf16* __restrict__ O) {      // [B][S][H*D]
  __shared__ __align__(16) __bf16 Qs[64][HEAD_DIM];        // 16 KB
  __shared__ __align__(16) __bf16 Ks[2][64][HEAD_DIM];     // 32 KB
  __shared__ __align__(16) __bf16 Vs[2][64][HEAD_DIM];     // 32 KB
  __shared__ __align__(16) __bf16 Ps[4][16][64];           //  8 KB

  const int tid  = threadIdx.x;
  const int lane = tid & 31;
  const int wave = tid >> 5;
  const int qb   = blockIdx.x;
  const int h    = blockIdx.y;
  const int b    = blockIdx.z;
  const int kv   = h >> 2;     // N_REP = 4

  const __bf16* Qg = Q + (((size_t)b * N_HEADS + h) * SEQ + (size_t)qb * 64) * HEAD_DIM;
  const __bf16* Kg = Kk + ((size_t)b * N_KV + kv) * SEQ * HEAD_DIM;
  const __bf16* Vg = Vv + ((size_t)b * N_KV + kv) * SEQ * HEAD_DIM;

  auto stageKV = [&](int buf, int j) {
#pragma unroll
    for (int i = 0; i < 8; ++i) {
      int idx = tid + i * 128;
      int row = idx >> 4;
      int seg = idx & 15;
      async_cp16(Kg + ((size_t)j * 64 + row) * HEAD_DIM + seg * 8, &Ks[buf][row][seg * 8]);
      async_cp16(Vg + ((size_t)j * 64 + row) * HEAD_DIM + seg * 8, &Vs[buf][row][seg * 8]);
    }
  };

  // Stage the 64x128 Q block once (synchronous; covered by first barrier).
#pragma unroll
  for (int i = 0; i < 8; ++i) {
    int idx = tid + i * 128;
    int row = idx >> 4;
    int seg = idx & 15;
    *(uint4*)(&Qs[row][seg * 8]) = *(const uint4*)(Qg + (size_t)row * HEAD_DIM + seg * 8);
  }
  stageKV(0, 0);

  v8f o[8];
#pragma unroll
  for (int i = 0; i < 8; ++i) o[i] = (v8f){0.f, 0.f, 0.f, 0.f, 0.f, 0.f, 0.f, 0.f};
  float rmax[8], rsum[8];
#pragma unroll
  for (int r = 0; r < 8; ++r) { rmax[r] = -3.0e38f; rsum[r] = 0.f; }

  const float scale = 0.08838834764831845f;   // 1/sqrt(128)
  const int   cn    = lane & 15;
  const int   half  = lane >> 4;

  for (int j = 0; j <= qb; ++j) {            // causal: stop at diagonal block
    const int cur = j & 1;
    if (j + 1 <= qb) {
      stageKV(cur ^ 1, j + 1);               // prefetch next K/V block
      wait_async<16>();                      // stage j's 16 ops complete
    } else {
      wait_async<0>();
    }
    __syncthreads();

    // S(16x64) = Q_wave(16x128) @ K_j(64x128)^T
    v8f sc[4];
#pragma unroll
    for (int nt = 0; nt < 4; ++nt) sc[nt] = (v8f){0.f, 0.f, 0.f, 0.f, 0.f, 0.f, 0.f, 0.f};
#pragma unroll
    for (int k0 = 0; k0 < HEAD_DIM; k0 += 32) {
      bf16x16 af = frag_a(&Qs[wave * 16][k0], HEAD_DIM, lane);
#pragma unroll
      for (int nt = 0; nt < 4; ++nt) {
        bf16x16 bfr = frag_b(&Ks[cur][nt * 16][k0], HEAD_DIM, lane);
        sc[nt] = __builtin_amdgcn_wmma_f32_16x16x32_bf16(
            false, af, false, bfr, (short)0, sc[nt], false, false);
      }
    }

    // Scale + causal mask (diagonal block only) + row-tile max.
    float tmax[8];
#pragma unroll
    for (int r = 0; r < 8; ++r) tmax[r] = -3.0e38f;
#pragma unroll
    for (int nt = 0; nt < 4; ++nt)
#pragma unroll
      for (int r = 0; r < 8; ++r) {
        float v = sc[nt][r] * scale;
        if (j == qb) {
          int kcol = j * 64 + nt * 16 + cn;
          int qrow = qb * 64 + wave * 16 + r + half * 8;
          if (kcol > qrow) v += -1.0e9f;
        }
        sc[nt][r] = v;
        tmax[r] = fmaxf(tmax[r], v);
      }
    // Reduce over the 16 lanes holding this row (xor stays inside the half).
#pragma unroll
    for (int r = 0; r < 8; ++r)
#pragma unroll
      for (int off = 1; off < 16; off <<= 1)
        tmax[r] = fmaxf(tmax[r], __shfl_xor(tmax[r], off, 32));

    // Online softmax update.
    float alpha[8], psum[8];
#pragma unroll
    for (int r = 0; r < 8; ++r) {
      float mnew = fmaxf(rmax[r], tmax[r]);
      alpha[r] = __expf(rmax[r] - mnew);
      rmax[r]  = mnew;
      psum[r]  = 0.f;
    }
#pragma unroll
    for (int nt = 0; nt < 4; ++nt)
#pragma unroll
      for (int r = 0; r < 8; ++r) {
        float p = __expf(sc[nt][r] - rmax[r]);
        sc[nt][r] = p;
        psum[r] += p;
      }
#pragma unroll
    for (int r = 0; r < 8; ++r) {
#pragma unroll
      for (int off = 1; off < 16; off <<= 1) psum[r] += __shfl_xor(psum[r], off, 32);
      rsum[r] = rsum[r] * alpha[r] + psum[r];
    }
#pragma unroll
    for (int dt = 0; dt < 8; ++dt)
#pragma unroll
      for (int r = 0; r < 8; ++r) o[dt][r] *= alpha[r];

    // Stage P (16x64) as bf16 in this wave's private LDS region.
#pragma unroll
    for (int nt = 0; nt < 4; ++nt)
#pragma unroll
      for (int r = 0; r < 8; ++r)
        Ps[wave][r + half * 8][nt * 16 + cn] = (__bf16)sc[nt][r];

    // O(16x128) += P(16x64) @ V_j(64x128)
#pragma unroll
    for (int k0 = 0; k0 < 64; k0 += 32) {
      bf16x16 af = frag_a(&Ps[wave][0][k0], 64, lane);
#pragma unroll
      for (int dt = 0; dt < 8; ++dt) {
        bf16x16 bfr = frag_b_kmajor(&Vs[cur][k0][dt * 16], HEAD_DIM, lane);
        o[dt] = __builtin_amdgcn_wmma_f32_16x16x32_bf16(
            false, af, false, bfr, (short)0, o[dt], false, false);
      }
    }
    __syncthreads();   // done reading Ks/Vs[cur]; prefetch may overwrite next iter
  }

  // Normalize and write [b][s][h*128 + d] as bf16 for the final projection.
  float inv[8];
#pragma unroll
  for (int r = 0; r < 8; ++r) inv[r] = 1.0f / rsum[r];
#pragma unroll
  for (int dt = 0; dt < 8; ++dt)
#pragma unroll
    for (int r = 0; r < 8; ++r) {
      int m  = wave * 16 + r + half * 8;
      int qi = qb * 64 + m;
      O[((size_t)b * SEQ + qi) * DIMV + (size_t)h * HEAD_DIM + dt * 16 + cn] =
          (__bf16)(o[dt][r] * inv[r]);
    }
}

// ---------------------------------------------------------------------------
// Workspace layout (bytes, all offsets 16B+ aligned)
// ---------------------------------------------------------------------------
static constexpr size_t XB_OFF  = 0;                                    // bf16 x        32M
static constexpr size_t WQB_OFF = XB_OFF  + MTOK * DIMV * 2;            // bf16 wq       32M
static constexpr size_t WKB_OFF = WQB_OFF + (size_t)DIMV * DIMV * 2;    // bf16 wk        8M
static constexpr size_t WVB_OFF = WKB_OFF + (size_t)KVDIM * DIMV * 2;   // bf16 wv        8M
static constexpr size_t WOB_OFF = WVB_OFF + (size_t)KVDIM * DIMV * 2;   // bf16 wo       32M
static constexpr size_t QF_OFF  = WOB_OFF + (size_t)DIMV * DIMV * 2;    // f32  Q        64M
static constexpr size_t KF_OFF  = QF_OFF  + MTOK * DIMV * 4;            // f32  K        16M
static constexpr size_t VF_OFF  = KF_OFF  + MTOK * KVDIM * 4;           // f32  V        16M
static constexpr size_t QR_OFF  = VF_OFF  + MTOK * KVDIM * 4;           // bf16 Q roped  32M
static constexpr size_t KR_OFF  = QR_OFF  + MTOK * DIMV * 2;            // bf16 K roped   8M
static constexpr size_t VB_OFF  = KR_OFF  + MTOK * KVDIM * 2;           // bf16 V packed  8M
static constexpr size_t AT_OFF  = VB_OFF  + MTOK * KVDIM * 2;           // bf16 attn out 32M

extern "C" void kernel_launch(void* const* d_in, const int* in_sizes, int n_in,
                              void* d_out, int out_size, void* d_ws, size_t ws_size,
                              hipStream_t stream) {
  (void)in_sizes; (void)n_in; (void)out_size; (void)ws_size;

  const float* x    = (const float*)d_in[0];
  const float* wq   = (const float*)d_in[1];
  const float* wk   = (const float*)d_in[2];
  const float* wv   = (const float*)d_in[3];
  const float* wo   = (const float*)d_in[4];
  const float* fcos = (const float*)d_in[5];
  const float* fsin = (const float*)d_in[6];
  // d_in[7] = mask (recomputed analytically), d_in[8] = start_pos (== 0)
  float* out = (float*)d_out;

  char* ws = (char*)d_ws;
  __bf16* xb  = (__bf16*)(ws + XB_OFF);
  __bf16* wqb = (__bf16*)(ws + WQB_OFF);
  __bf16* wkb = (__bf16*)(ws + WKB_OFF);
  __bf16* wvb = (__bf16*)(ws + WVB_OFF);
  __bf16* wob = (__bf16*)(ws + WOB_OFF);
  float*  qf  = (float*)(ws + QF_OFF);
  float*  kf  = (float*)(ws + KF_OFF);
  float*  vf  = (float*)(ws + VF_OFF);
  __bf16* qr  = (__bf16*)(ws + QR_OFF);
  __bf16* kr  = (__bf16*)(ws + KR_OFF);
  __bf16* vb  = (__bf16*)(ws + VB_OFF);
  __bf16* at  = (__bf16*)(ws + AT_OFF);

  const size_t nx  = MTOK * DIMV;             // 16M
  const size_t nwq = (size_t)DIMV * DIMV;     // 16M
  const size_t nwk = (size_t)KVDIM * DIMV;    //  4M

  // 1) downconvert inputs to bf16
  cvt_f32_bf16<<<dim3((unsigned)(nx  / 1024)), 256, 0, stream>>>(x,  xb,  nx);
  cvt_f32_bf16<<<dim3((unsigned)(nwq / 1024)), 256, 0, stream>>>(wq, wqb, nwq);
  cvt_f32_bf16<<<dim3((unsigned)(nwk / 1024)), 256, 0, stream>>>(wk, wkb, nwk);
  cvt_f32_bf16<<<dim3((unsigned)(nwk / 1024)), 256, 0, stream>>>(wv, wvb, nwk);
  cvt_f32_bf16<<<dim3((unsigned)(nwq / 1024)), 256, 0, stream>>>(wo, wob, nwq);

  // 2) QKV projections (WMMA GEMMs): C[M,N] = A[M,K] * B[N,K]^T
  gemm_bf16_f32<<<dim3(DIMV / 128, (unsigned)(MTOK / 128)), 256, 0, stream>>>(
      xb, wqb, qf, (int)MTOK, DIMV, DIMV);
  gemm_bf16_f32<<<dim3(KVDIM / 128, (unsigned)(MTOK / 128)), 256, 0, stream>>>(
      xb, wkb, kf, (int)MTOK, KVDIM, DIMV);
  gemm_bf16_f32<<<dim3(KVDIM / 128, (unsigned)(MTOK / 128)), 256, 0, stream>>>(
      xb, wvb, vf, (int)MTOK, KVDIM, DIMV);

  // 3) RoPE + transpose to [b][head][s][d] bf16
  const size_t nq_pairs = (size_t)BSZ * SEQ * N_HEADS * 64;  // 8.4M
  const size_t nk_pairs = (size_t)BSZ * SEQ * N_KV * 64;     // 2.1M
  const size_t nv_elems = (size_t)BSZ * SEQ * N_KV * HEAD_DIM;
  rope_pack<<<dim3((unsigned)(nq_pairs / 256)), 256, 0, stream>>>(qf, fcos, fsin, qr, N_HEADS);
  rope_pack<<<dim3((unsigned)(nk_pairs / 256)), 256, 0, stream>>>(kf, fcos, fsin, kr, N_KV);
  v_pack<<<dim3((unsigned)(nv_elems / 256)), 256, 0, stream>>>(vf, vb);

  // 4) fused causal GQA flash attention
  flash_attn<<<dim3(SEQ / 64, N_HEADS, BSZ), 128, 0, stream>>>(qr, kr, vb, at);

  // 5) output projection straight into d_out (f32)
  gemm_bf16_f32<<<dim3(DIMV / 128, (unsigned)(MTOK / 128)), 256, 0, stream>>>(
      at, wob, out, (int)MTOK, DIMV, DIMV);
}
